// Attention_44281112822191
// MI455X (gfx1250) — compile-verified, matrix operand order checked by reading, not run on hardware
//
#include <hip/hip_runtime.h>
#include <hip/hip_bf16.h>

typedef __attribute__((ext_vector_type(16))) __bf16 v16bf;
typedef __attribute__((ext_vector_type(8)))  __bf16 v8bf;
typedef __attribute__((ext_vector_type(8)))  float  v8f;

constexpr int B_   = 2;
constexpr int S_   = 2048;
constexpr int E_   = 2048;
constexpr int H_   = 32;
constexpr int HKV_ = 8;
constexpr int D_   = 64;
constexpr int BS_  = B_ * S_;      // 4096 rows

// ---------------------------------------------------------------------------
// CDNA5 async global->LDS helpers (ASYNCcnt path, bypasses VGPRs)
// ---------------------------------------------------------------------------
__device__ __forceinline__ unsigned lds_off(const void* p) {
  // generic -> LDS(AS3) addrspacecast, then take the 32-bit LDS byte offset
  return (unsigned)(unsigned long long)(const __attribute__((address_space(3))) void*)p;
}

__device__ __forceinline__ void a2l_b128(unsigned dst_lds, const void* src) {
  asm volatile("global_load_async_to_lds_b128 %0, %1, off"
               :: "v"(dst_lds), "v"(src) : "memory");
}

__device__ __forceinline__ void wait_async0() {
  asm volatile("s_wait_asynccnt 0x0" ::: "memory");
}

// ---------------------------------------------------------------------------
// WMMA helpers
// ---------------------------------------------------------------------------
__device__ __forceinline__ v8f wmma_bf(v16bf a, v16bf b, v8f c) {
  return __builtin_amdgcn_wmma_f32_16x16x32_bf16(
      /*neg_a=*/false, a, /*neg_b=*/false, b,
      /*c_mod=*/(short)0, c, /*reuse_a=*/false, /*reuse_b=*/false);
}

// 16x32 bf16 fragment from LDS in the documented A/B VGPR layout:
// lane m = lane&15 -> row, half = lane>>4 -> K offset;
// element e -> k = (e<8 ? 0 : 16) + half*8 + (e&7).  Two 16B ds loads.
__device__ __forceinline__ v16bf ld_frag(const __bf16* base, int row0, int col0, int ld) {
  int lane = threadIdx.x & 31;
  int hf   = (lane >> 4) & 1;
  int m    = lane & 15;
  const __bf16* p = base + (row0 + m) * ld + col0 + hf * 8;
  v8bf lo = *(const v8bf*)(p);
  v8bf hi = *(const v8bf*)(p + 16);
  v16bf f;
#pragma unroll
  for (int e = 0; e < 8; ++e) { f[e] = lo[e]; f[e + 8] = hi[e]; }
  return f;
}

// ---------------------------------------------------------------------------
// small utility kernels
// ---------------------------------------------------------------------------
__global__ void fill_zero_f32(float* __restrict__ p, int n) {
  int i = blockIdx.x * blockDim.x + threadIdx.x;
  if (i < n) p[i] = 0.0f;
}

__global__ void cvt_f32_bf16(const float* __restrict__ src, __bf16* __restrict__ dst, long n) {
  long i = (long)blockIdx.x * blockDim.x + threadIdx.x;
  if (i < n) dst[i] = (__bf16)src[i];
}

// Batched transpose+convert: dst[z][c][r] = (bf16) src[off(z) + r*rs + c]
// off(z) = (z/n1)*sb0 + (z%n1)*sb1.  R, C multiples of 32.  256 threads.
__global__ __launch_bounds__(256)
void transp_cvt(const float* __restrict__ src, __bf16* __restrict__ dst,
                int R, int C, long rs, long sb0, long sb1, int n1) {
  __shared__ float t[32][33];
  int z  = blockIdx.z;
  const float* s = src + (long)(z / n1) * sb0 + (long)(z % n1) * sb1;
  __bf16* d = dst + (long)z * R * C;
  int r0 = blockIdx.y * 32, c0 = blockIdx.x * 32;
  int tx = threadIdx.x & 31, ty = threadIdx.x >> 5;   // 32 x 8
#pragma unroll
  for (int u = 0; u < 4; ++u)
    t[ty + u * 8][tx] = s[(long)(r0 + ty + u * 8) * rs + c0 + tx];
  __syncthreads();
#pragma unroll
  for (int u = 0; u < 4; ++u)
    d[(long)(c0 + ty + u * 8) * R + r0 + tx] = (__bf16)t[tx][ty + u * 8];
}

// RoPE + f32->bf16 (scale folds 1/sqrt(D) into q). src: (B*S, nh, 64); cos/sin: (S,32)
__global__ void rope_cvt(const float* __restrict__ src, const float* __restrict__ cosb,
                         const float* __restrict__ sinb, __bf16* __restrict__ dst,
                         int nheads, float scale, long npairs) {
  long i = (long)blockIdx.x * blockDim.x + threadIdx.x;
  if (i >= npairs) return;
  int  dp   = (int)(i & 31);
  long rest = i >> 5;
  int  h    = (int)(rest % nheads);
  long bs   = rest / nheads;
  int  s    = (int)(bs % S_);
  float c  = cosb[(long)s * 32 + dp];
  float sn = sinb[(long)s * 32 + dp];
  long base = (bs * nheads + h) * D_ + dp * 2;
  float x0 = src[base], x1 = src[base + 1];
  dst[base]     = (__bf16)((x0 * c - x1 * sn) * scale);
  dst[base + 1] = (__bf16)((x0 * sn + x1 * c) * scale);
}

// ---------------------------------------------------------------------------
// bf16 GEMM: C(f32, MxN) = A(bf16 MxK row-major) * Bt(bf16 NxK row-major)^T + bias
// 128x128x32 block, 256 threads = 8 waves (wave tile 64x32 = 4x2 WMMAs).
// Double-buffered LDS, all staging via global_load_async_to_lds_b128.
// ---------------------------------------------------------------------------
#define GM 128
#define GN 128
#define GK 32
#define LDT 40   // GK + 8 pad; 80B stride = multiple of 16B

__global__ __launch_bounds__(256)
void gemm_bf16(const __bf16* __restrict__ A, const __bf16* __restrict__ Bt,
               float* __restrict__ C, const float* __restrict__ bias,
               int M, int N, int K) {
  __shared__ __bf16 As[2][GM * LDT];
  __shared__ __bf16 Bs[2][GN * LDT];

  int tid  = threadIdx.x;
  int lane = tid & 31;
  int wv   = tid >> 5;
  int hf   = (lane >> 4) & 1;
  int ln   = lane & 15;
  int bm0  = blockIdx.y * GM;
  int bn0  = blockIdx.x * GN;
  int wm   = (wv >> 2) * 64;
  int wn   = (wv & 3) * 32;

  v8f acc[4][2];
#pragma unroll
  for (int i = 0; i < 4; ++i)
#pragma unroll
    for (int j = 0; j < 2; ++j) acc[i][j] = (v8f){};

  // per thread: one 128-row tile row, 16 contiguous k (two b128 chunks) for A and Bt
  int sr = tid >> 1, scc = (tid & 1) * 16;
  auto stage = [&](int buf, int k0) {
    const __bf16* ga = A  + (size_t)(bm0 + sr) * K + k0 + scc;
    a2l_b128(lds_off(&As[buf][sr * LDT + scc]),     ga);
    a2l_b128(lds_off(&As[buf][sr * LDT + scc + 8]), ga + 8);
    const __bf16* gb = Bt + (size_t)(bn0 + sr) * K + k0 + scc;
    a2l_b128(lds_off(&Bs[buf][sr * LDT + scc]),     gb);
    a2l_b128(lds_off(&Bs[buf][sr * LDT + scc + 8]), gb + 8);
  };

  int nk = K / GK;
  stage(0, 0);
  for (int it = 0; it < nk; ++it) {
    wait_async0();
    __syncthreads();                 // tile `it` fully in LDS, all waves done with prev buf
    if (it + 1 < nk) stage((it + 1) & 1, (it + 1) * GK);
    const __bf16* as = As[it & 1];
    const __bf16* bs = Bs[it & 1];

    v16bf afr[4], bfr[2];
#pragma unroll
    for (int i = 0; i < 4; ++i) afr[i] = ld_frag(as, wm + i * 16, 0, LDT);
#pragma unroll
    for (int j = 0; j < 2; ++j) bfr[j] = ld_frag(bs, wn + j * 16, 0, LDT);
#pragma unroll
    for (int i = 0; i < 4; ++i)
#pragma unroll
      for (int j = 0; j < 2; ++j)
        acc[i][j] = wmma_bf(afr[i], bfr[j], acc[i][j]);
  }

#pragma unroll
  for (int i = 0; i < 4; ++i)
#pragma unroll
    for (int j = 0; j < 2; ++j) {
      int row = bm0 + wm + i * 16 + hf * 8;
      int col = bn0 + wn + j * 16 + ln;
      float bv = bias[col];
#pragma unroll
      for (int r = 0; r < 8; ++r)
        C[(size_t)(row + r) * N + col] = acc[i][j][r] + bv;
    }
}

// ---------------------------------------------------------------------------
// Flash attention (non-causal, start_pos=0, GQA 32/8).
// Block: 128 q rows x 1 head; 256 threads = 8 waves x 16 rows.
// q pre-scaled by 1/sqrt(D).  K/V tiles async-staged + double-buffered.
// V comes pre-transposed from global: vt[b][hkv][d][s].
// ---------------------------------------------------------------------------
#define AQ 128
#define AK 128
#define LDQ 72    // 64 + 8 pad
#define LDP 136   // 128 + 8 pad

__global__ __launch_bounds__(256)
void attn_fwd(const __bf16* __restrict__ qb, const __bf16* __restrict__ kb,
              const __bf16* __restrict__ vt, __bf16* __restrict__ ob) {
  __shared__ __bf16 Qs[AQ * LDQ];       // q rows x d
  __shared__ __bf16 Ks[2][AK * LDQ];    // key rows x d
  __shared__ __bf16 Vs[2][D_ * LDP];    // d rows x key
  __shared__ __bf16 Ps[AQ * LDP];       // q rows x key

  int tid  = threadIdx.x;
  int lane = tid & 31;
  int wv   = tid >> 5;
  int hf   = (lane >> 4) & 1;
  int ln   = lane & 15;
  int q0   = blockIdx.x * AQ;
  int h    = blockIdx.y;
  int b    = blockIdx.z;
  int hk   = h >> 2;

  const __bf16* qptr = qb + (((long)b * S_ + q0) * H_ + h) * D_;
  const __bf16* kbb  = kb + ((long)b * S_ * HKV_ + hk) * D_;
  const __bf16* vtp  = vt + ((long)(b * HKV_ + hk) * D_) * S_;

  { // async-stage Q tile (128 x 64): 1024 b128 chunks / 256 threads
#pragma unroll
    for (int u = 0; u < 4; ++u) {
      int cid = tid + 256 * u;
      int r = cid >> 3, c = (cid & 7) * 8;
      a2l_b128(lds_off(&Qs[r * LDQ + c]), qptr + (long)r * (H_ * D_) + c);
    }
  }
  auto stage_kv = [&](int buf, int kt) {
#pragma unroll
    for (int u = 0; u < 4; ++u) {       // K tile: 128 keys x 64 d
      int cid = tid + 256 * u;
      int r = cid >> 3, c = (cid & 7) * 8;
      a2l_b128(lds_off(&Ks[buf][r * LDQ + c]),
               kbb + (long)(kt + r) * (HKV_ * D_) + c);
    }
#pragma unroll
    for (int u = 0; u < 4; ++u) {       // V tile (pre-transposed): 64 d x 128 keys
      int cid = tid + 256 * u;
      int d = cid >> 4, c = (cid & 15) * 8;
      a2l_b128(lds_off(&Vs[buf][d * LDP + c]), vtp + (long)d * S_ + kt + c);
    }
  };

  v8f oacc[4];
#pragma unroll
  for (int dt = 0; dt < 4; ++dt) oacc[dt] = (v8f){};
  float m_run[8], l_run[8];
#pragma unroll
  for (int r = 0; r < 8; ++r) { m_run[r] = -1e30f; l_run[r] = 0.0f; }

  stage_kv(0, 0);
  constexpr int NT = S_ / AK;
  for (int it = 0; it < NT; ++it) {
    wait_async0();
    __syncthreads();                    // Q + K/V tile `it` resident; prev PV done
    if (it + 1 < NT) stage_kv((it + 1) & 1, (it + 1) * AK);
    const __bf16* ks = Ks[it & 1];
    const __bf16* vs = Vs[it & 1];

    // ---- scores: S(16 x 128) = Q(16x64) * K^T, two 32-wide k chunks
    v16bf aq0 = ld_frag(Qs, wv * 16, 0,  LDQ);
    v16bf aq1 = ld_frag(Qs, wv * 16, 32, LDQ);
    float sc[8][8];
#pragma unroll
    for (int nt = 0; nt < 8; ++nt) {
      v16bf bk0 = ld_frag(ks, nt * 16, 0,  LDQ);
      v16bf bk1 = ld_frag(ks, nt * 16, 32, LDQ);
      v8f c0 = (v8f){};
      c0 = wmma_bf(aq0, bk0, c0);
      c0 = wmma_bf(aq1, bk1, c0);
#pragma unroll
      for (int r = 0; r < 8; ++r) sc[nt][r] = c0[r];
    }

    // ---- online softmax (C-frag rows: M = hf*8 + r, cols in 16-lane halves)
    float mnew[8], scl[8], rs[8];
#pragma unroll
    for (int r = 0; r < 8; ++r) {
      float mx = sc[0][r];
#pragma unroll
      for (int nt = 1; nt < 8; ++nt) mx = fmaxf(mx, sc[nt][r]);
#pragma unroll
      for (int m = 1; m < 16; m <<= 1) mx = fmaxf(mx, __shfl_xor(mx, m, 32));
      mnew[r] = fmaxf(m_run[r], mx);
      scl[r]  = __expf(m_run[r] - mnew[r]);
      rs[r]   = 0.0f;
    }
#pragma unroll
    for (int nt = 0; nt < 8; ++nt)
#pragma unroll
      for (int r = 0; r < 8; ++r) {
        float p = __expf(sc[nt][r] - mnew[r]);
        rs[r] += p;
        Ps[(wv * 16 + hf * 8 + r) * LDP + nt * 16 + ln] = (__bf16)p;
      }
#pragma unroll
    for (int r = 0; r < 8; ++r) {
      float t = rs[r];
#pragma unroll
      for (int m = 1; m < 16; m <<= 1) t += __shfl_xor(t, m, 32);
      l_run[r] = l_run[r] * scl[r] + t;
      m_run[r] = mnew[r];
    }
#pragma unroll
    for (int dt = 0; dt < 4; ++dt)
#pragma unroll
      for (int r = 0; r < 8; ++r) oacc[dt][r] *= scl[r];

    __syncthreads();                    // P fully written before PV

    // ---- O(16x64) += P(16x128) * V(128x64), 4 k-chunks of 32 keys
#pragma unroll
    for (int kc = 0; kc < 4; ++kc) {
      v16bf ap = ld_frag(Ps, wv * 16, kc * 32, LDP);
#pragma unroll
      for (int dt = 0; dt < 4; ++dt) {
        v16bf bv = ld_frag(vs, dt * 16, kc * 32, LDP);
        oacc[dt] = wmma_bf(ap, bv, oacc[dt]);
      }
    }
  }

  // ---- epilogue: normalize, write (B,S,H,D) bf16
#pragma unroll
  for (int dt = 0; dt < 4; ++dt)
#pragma unroll
    for (int r = 0; r < 8; ++r) {
      int row = q0 + wv * 16 + hf * 8 + r;
      int d   = dt * 16 + ln;
      float val = oacc[dt][r] / l_run[r];
      ob[(((long)b * S_ + row) * H_ + h) * D_ + d] = (__bf16)val;
    }
}

// ---------------------------------------------------------------------------
// Host launch
// ---------------------------------------------------------------------------
extern "C" void kernel_launch(void* const* d_in, const int* in_sizes, int n_in,
                              void* d_out, int out_size, void* d_ws, size_t ws_size,
                              hipStream_t stream) {
  const float* x     = (const float*)d_in[0];
  const float* fcos  = (const float*)d_in[2];
  const float* fsin  = (const float*)d_in[3];
  const float* wq    = (const float*)d_in[4];
  const float* wk    = (const float*)d_in[5];
  const float* wv    = (const float*)d_in[6];
  const float* out_w = (const float*)d_in[7];
  const float* out_b = (const float*)d_in[8];
  float* out = (float*)d_out;

  size_t off = 0;
  auto take = [&](size_t bytes) -> char* {
    char* p = (char*)d_ws + off;
    off += (bytes + 255) & ~(size_t)255;
    return p;
  };
  __bf16* x_bf = (__bf16*)take((size_t)BS_ * E_ * 2);
  __bf16* wqt  = (__bf16*)take((size_t)E_ * H_ * D_ * 2);     // (N=2048) x (K=2048)
  __bf16* wkt  = (__bf16*)take((size_t)E_ * HKV_ * D_ * 2);   // 512 x 2048
  __bf16* wvt  = (__bf16*)take((size_t)E_ * HKV_ * D_ * 2);
  __bf16* owt  = (__bf16*)take((size_t)H_ * D_ * E_ * 2);     // 2048 x 2048
  float*  q_f  = (float*) take((size_t)BS_ * H_ * D_ * 4);
  float*  k_f  = (float*) take((size_t)BS_ * HKV_ * D_ * 4);
  float*  v_f  = (float*) take((size_t)BS_ * HKV_ * D_ * 4);
  __bf16* q_bf = (__bf16*)take((size_t)BS_ * H_ * D_ * 2);
  __bf16* k_bf = (__bf16*)take((size_t)BS_ * HKV_ * D_ * 2);
  __bf16* vt   = (__bf16*)take((size_t)BS_ * HKV_ * D_ * 2);  // (b,hkv,d,s)
  float*  zb   = (float*) take((size_t)E_ * 4);
  __bf16* o_bf = (__bf16*)q_f;    // q_f dead after RoPE

  fill_zero_f32<<<(E_ + 255) / 256, 256, 0, stream>>>(zb, E_);
  cvt_f32_bf16<<<(int)(((long)BS_ * E_ + 255) / 256), 256, 0, stream>>>(x, x_bf, (long)BS_ * E_);

  // weight transposes: dst[n][k] = w[k][n]
  transp_cvt<<<dim3(E_ / 32, E_ / 32, 1), 256, 0, stream>>>(wq, wqt, E_, E_, E_, 0, 0, 1);
  transp_cvt<<<dim3((HKV_ * D_) / 32, E_ / 32, 1), 256, 0, stream>>>(wk, wkt, E_, HKV_ * D_, HKV_ * D_, 0, 0, 1);
  transp_cvt<<<dim3((HKV_ * D_) / 32, E_ / 32, 1), 256, 0, stream>>>(wv, wvt, E_, HKV_ * D_, HKV_ * D_, 0, 0, 1);
  transp_cvt<<<dim3(E_ / 32, E_ / 32, 1), 256, 0, stream>>>(out_w, owt, H_ * D_, E_, E_, 0, 0, 1);

  // QKV projections
  gemm_bf16<<<dim3((H_ * D_) / GN, BS_ / GM), 256, 0, stream>>>(x_bf, wqt, q_f, zb, BS_, H_ * D_, E_);
  gemm_bf16<<<dim3((HKV_ * D_) / GN, BS_ / GM), 256, 0, stream>>>(x_bf, wkt, k_f, zb, BS_, HKV_ * D_, E_);
  gemm_bf16<<<dim3((HKV_ * D_) / GN, BS_ / GM), 256, 0, stream>>>(x_bf, wvt, v_f, zb, BS_, HKV_ * D_, E_);

  // RoPE (+1/sqrt(D) into q), V transposed to (b,hkv,d,s)
  {
    long npq = (long)BS_ * H_ * (D_ / 2);
    rope_cvt<<<(int)((npq + 255) / 256), 256, 0, stream>>>(q_f, fcos, fsin, q_bf, H_, 0.125f, npq);
    long npk = (long)BS_ * HKV_ * (D_ / 2);
    rope_cvt<<<(int)((npk + 255) / 256), 256, 0, stream>>>(k_f, fcos, fsin, k_bf, HKV_, 1.0f, npk);
    // v_f (b,s,hkv,d) -> vt (b,hkv,d,s): per (b,hkv) slice R=S rows, C=D cols
    transp_cvt<<<dim3(D_ / 32, S_ / 32, B_ * HKV_), 256, 0, stream>>>(
        v_f, vt, S_, D_, (long)HKV_ * D_, (long)S_ * HKV_ * D_, (long)D_, HKV_);
  }

  // flash attention
  attn_fwd<<<dim3(S_ / AQ, H_, B_), 256, 0, stream>>>(q_bf, k_bf, vt, o_bf);

  // output projection + bias
  gemm_bf16<<<dim3(E_ / GN, BS_ / GM), 256, 0, stream>>>(o_bf, owt, out, out_b, BS_, E_, H_ * D_);
}
